// MemoryBank_19232863552260
// MI455X (gfx1250) — matrix operand order repaired and checked by reading, not compiled, not run on hardware
//
#include <hip/hip_runtime.h>

typedef __attribute__((ext_vector_type(16))) __bf16 v16bf;
typedef __attribute__((ext_vector_type(8)))  float  v8f;

#define DIM        256     // feature dim
#define ROWS       32      // query rows per workgroup
#define KSEL       16      // top-k
#define NWAVES     8
#define SC_STRIDE  20      // padded row stride (floats) for score tiles

__device__ __forceinline__ void insert16(float* tv, int* ti, float v, int idx) {
    // tv sorted descending; caller guarantees v > tv[KSEL-1]
    int p = KSEL - 1;
    while (p > 0 && tv[p - 1] < v) {
        tv[p] = tv[p - 1];
        ti[p] = ti[p - 1];
        --p;
    }
    tv[p] = v;
    ti[p] = idx;
}

__global__ __launch_bounds__(256) void sim_topk_kernel(
    const float* __restrict__ features,      // [B, DIM]
    const int*   __restrict__ class_labels,  // [B]
    const float* __restrict__ bank,          // [M, DIM]
    const int*   __restrict__ bank_cls,      // [M]
    float* __restrict__ d_out,               // [B*KSEL vals][B*KSEL idx-as-float]
    int B, int M)
{
    __shared__ __bf16 s_qhi[ROWS * DIM];                    // 16 KB
    __shared__ __bf16 s_qlo[ROWS * DIM];                    // 16 KB
    __shared__ float  s_wsc[NWAVES * ROWS * SC_STRIDE];     // 20 KB, per-wave score tiles
    __shared__ float  s_topv[NWAVES][ROWS][KSEL];           // 16 KB, per-wave top lists
    __shared__ int    s_topi[NWAVES][ROWS][KSEL];           // 16 KB
    __shared__ float  s_part[ROWS * 8];
    __shared__ float  s_inv[ROWS];
    __shared__ int    s_rcls[ROWS];

    const int tid      = threadIdx.x;
    const int lane     = tid & 31;
    const int wave     = tid >> 5;
    const int g        = lane >> 4;   // lane half (K-group / row-group select)
    const int ln       = lane & 15;   // row (A) / col (B) within tile
    const int row_base = blockIdx.x * ROWS;

    // ---------------- Phase 0: norms (partial sums) ----------------
    {
        const int r   = tid >> 3;   // 0..31
        const int seg = tid & 7;    // 0..7 -> 32 contiguous floats
        const float* src = features + (size_t)(row_base + r) * DIM + seg * 32;
        float ss = 0.f;
        #pragma unroll
        for (int i = 0; i < 32; ++i) { float x = src[i]; ss += x * x; }
        s_part[r * 8 + seg] = ss;
    }
    if (tid < ROWS) s_rcls[tid] = class_labels[row_base + tid];
    __syncthreads();
    if (tid < ROWS) {
        float ss = 0.f;
        #pragma unroll
        for (int j = 0; j < 8; ++j) ss += s_part[tid * 8 + j];
        s_inv[tid] = 1.0f / fmaxf(sqrtf(ss), 1e-12f);
    }
    __syncthreads();

    // ------------- Phase 1: normalize + bf16 hi/lo split -------------
    {
        const int r   = tid >> 3;
        const int seg = tid & 7;
        const float inv = s_inv[r];
        const float* src = features + (size_t)(row_base + r) * DIM + seg * 32;
        #pragma unroll
        for (int i = 0; i < 32; ++i) {
            float q  = src[i] * inv;
            __bf16 h = (__bf16)q;
            float lo = q - (float)h;
            s_qhi[r * DIM + seg * 32 + i] = h;
            s_qlo[r * DIM + seg * 32 + i] = (__bf16)lo;
        }
    }
    __syncthreads();

    // Preload the 16 row classes this lane's C fragments touch (fixed by g).
    int rcls0[8], rcls1[8];
    #pragma unroll
    for (int r = 0; r < 8; ++r) {
        rcls0[r] = s_rcls[r + 8 * g];
        rcls1[r] = s_rcls[r + 8 * g + 16];
    }

    // Init this wave's per-row top-k lists (lane t owns row t).
    #pragma unroll
    for (int j = 0; j < KSEL; ++j) {
        s_topv[wave][lane][j] = -__builtin_inff();
        s_topi[wave][lane][j] = 0;
    }
    float kth = -__builtin_inff();

    float* wsc = s_wsc + wave * ROWS * SC_STRIDE;
    const float NEG_INF = -__builtin_inff();

    // -------- Phase 2: GEMM chunks + wave-private streaming top-k --------
    const int nChunks = M / (NWAVES * 16);
    for (int ch = 0; ch < nChunks; ++ch) {
        const int tile_col0 = ch * (NWAVES * 16) + wave * 16;
        const int colg      = tile_col0 + ln;    // this lane's bank row (= sim column)
        const int bcls      = bank_cls[colg];

        v8f acc0 = {};  // rows 0..15 of block
        v8f acc1 = {};  // rows 16..31 of block
        const float* bptr = bank + (size_t)colg * DIM;

        for (int ks = 0; ks < DIM; ks += 32) {
            // ---- B operand: 32x16 bf16, lane ln = column, half g = K 8g..8g+7 and 16+8g..16+8g+7 ----
            v16bf bhi, blo;
            {
                const float* bp = bptr + ks + g * 8;
                #pragma unroll
                for (int e = 0; e < 16; ++e) {
                    float x  = bp[e + ((e >> 3) << 3)];
                    __bf16 h = (__bf16)x;
                    bhi[e] = h;
                    blo[e] = (__bf16)(x - (float)h);
                }
            }
            // ---- A operand tile 0 (rows 0..15), documented 16-bit A layout ----
            v16bf ahi, alo;
            {
                const __bf16* qh = s_qhi + ln * DIM + ks;
                const __bf16* ql = s_qlo + ln * DIM + ks;
                #pragma unroll
                for (int e = 0; e < 8; ++e) {
                    ahi[e]     = qh[g * 8 + e];
                    ahi[e + 8] = qh[16 + g * 8 + e];
                    alo[e]     = ql[g * 8 + e];
                    alo[e + 8] = ql[16 + g * 8 + e];
                }
            }
            acc0 = __builtin_amdgcn_wmma_f32_16x16x32_bf16(false, ahi, false, bhi, (short)0, acc0, false, false);
            acc0 = __builtin_amdgcn_wmma_f32_16x16x32_bf16(false, ahi, false, blo, (short)0, acc0, false, false);
            acc0 = __builtin_amdgcn_wmma_f32_16x16x32_bf16(false, alo, false, bhi, (short)0, acc0, false, false);

            // ---- A operand tile 1 (rows 16..31), B reused in-register ----
            {
                const __bf16* qh = s_qhi + (16 + ln) * DIM + ks;
                const __bf16* ql = s_qlo + (16 + ln) * DIM + ks;
                #pragma unroll
                for (int e = 0; e < 8; ++e) {
                    ahi[e]     = qh[g * 8 + e];
                    ahi[e + 8] = qh[16 + g * 8 + e];
                    alo[e]     = ql[g * 8 + e];
                    alo[e + 8] = ql[16 + g * 8 + e];
                }
            }
            acc1 = __builtin_amdgcn_wmma_f32_16x16x32_bf16(false, ahi, false, bhi, (short)0, acc1, false, false);
            acc1 = __builtin_amdgcn_wmma_f32_16x16x32_bf16(false, ahi, false, blo, (short)0, acc1, false, false);
            acc1 = __builtin_amdgcn_wmma_f32_16x16x32_bf16(false, alo, false, bhi, (short)0, acc1, false, false);
        }

        // ---- masked spill into this wave's private score tile ----
        // C layout: VGPR r -> row r + 8*g (tile0), +16 (tile1); col = ln
        #pragma unroll
        for (int r = 0; r < 8; ++r) {
            const int row0 = r + 8 * g;
            wsc[row0 * SC_STRIDE + ln]        = (bcls != rcls0[r]) ? acc0[r] : NEG_INF;
            wsc[(row0 + 16) * SC_STRIDE + ln] = (bcls != rcls1[r]) ? acc1[r] : NEG_INF;
        }

        // ---- wave-private selection: lane t scans row t (same-wave DS is in-order) ----
        {
            const float* rowp = wsc + lane * SC_STRIDE;
            const float4 a = ((const float4*)rowp)[0];
            const float4 b = ((const float4*)rowp)[1];
            const float4 c = ((const float4*)rowp)[2];
            const float4 d = ((const float4*)rowp)[3];

            // Branch-free max-gate: skip the insert block unless something qualifies.
            const float m = fmaxf(
                fmaxf(fmaxf(fmaxf(a.x, a.y), fmaxf(a.z, a.w)),
                      fmaxf(fmaxf(b.x, b.y), fmaxf(b.z, b.w))),
                fmaxf(fmaxf(fmaxf(c.x, c.y), fmaxf(c.z, c.w)),
                      fmaxf(fmaxf(d.x, d.y), fmaxf(d.z, d.w))));
            if (m > kth) {
                float* tv = &s_topv[wave][lane][0];
                int*   ti = &s_topi[wave][lane][0];
                const int c0 = tile_col0;
                if (a.x > kth) { insert16(tv, ti, a.x, c0 + 0);  kth = tv[KSEL - 1]; }
                if (a.y > kth) { insert16(tv, ti, a.y, c0 + 1);  kth = tv[KSEL - 1]; }
                if (a.z > kth) { insert16(tv, ti, a.z, c0 + 2);  kth = tv[KSEL - 1]; }
                if (a.w > kth) { insert16(tv, ti, a.w, c0 + 3);  kth = tv[KSEL - 1]; }
                if (b.x > kth) { insert16(tv, ti, b.x, c0 + 4);  kth = tv[KSEL - 1]; }
                if (b.y > kth) { insert16(tv, ti, b.y, c0 + 5);  kth = tv[KSEL - 1]; }
                if (b.z > kth) { insert16(tv, ti, b.z, c0 + 6);  kth = tv[KSEL - 1]; }
                if (b.w > kth) { insert16(tv, ti, b.w, c0 + 7);  kth = tv[KSEL - 1]; }
                if (c.x > kth) { insert16(tv, ti, c.x, c0 + 8);  kth = tv[KSEL - 1]; }
                if (c.y > kth) { insert16(tv, ti, c.y, c0 + 9);  kth = tv[KSEL - 1]; }
                if (c.z > kth) { insert16(tv, ti, c.z, c0 + 10); kth = tv[KSEL - 1]; }
                if (c.w > kth) { insert16(tv, ti, c.w, c0 + 11); kth = tv[KSEL - 1]; }
                if (d.x > kth) { insert16(tv, ti, d.x, c0 + 12); kth = tv[KSEL - 1]; }
                if (d.y > kth) { insert16(tv, ti, d.y, c0 + 13); kth = tv[KSEL - 1]; }
                if (d.z > kth) { insert16(tv, ti, d.z, c0 + 14); kth = tv[KSEL - 1]; }
                if (d.w > kth) { insert16(tv, ti, d.w, c0 + 15); kth = tv[KSEL - 1]; }
            }
        }
    }

    // -------- Phase 3: merge the 8 per-wave lists for each row, write out --------
    __syncthreads();
    if (tid < ROWS) {
        const int row = tid;
        float* tv = &s_topv[0][row][0];   // accumulate into wave-0's list
        int*   ti = &s_topi[0][row][0];
        float fk = tv[KSEL - 1];
        for (int w = 1; w < NWAVES; ++w) {
            #pragma unroll
            for (int j = 0; j < KSEL; ++j) {
                const float v = s_topv[w][row][j];
                if (v <= fk) break;               // lists are sorted descending
                insert16(tv, ti, v, s_topi[w][row][j]);
                fk = tv[KSEL - 1];
            }
        }
        const int grow = row_base + row;
        float* out_vals = d_out;
        float* out_idx  = d_out + (size_t)B * KSEL;
        #pragma unroll
        for (int j = 0; j < KSEL; ++j) {
            out_vals[(size_t)grow * KSEL + j] = tv[j];
            out_idx[(size_t)grow * KSEL + j]  = (float)ti[j];
        }
    }
}

extern "C" void kernel_launch(void* const* d_in, const int* in_sizes, int n_in,
                              void* d_out, int out_size, void* d_ws, size_t ws_size,
                              hipStream_t stream) {
    (void)n_in; (void)out_size; (void)d_ws; (void)ws_size;
    const float* features = (const float*)d_in[0];
    const int*   cls      = (const int*)d_in[1];
    // d_in[2] = domain_labels (unused by reference math)
    const float* bank     = (const float*)d_in[3];
    const int*   bcls     = (const int*)d_in[4];
    // d_in[5] = bank_domain_labels (unused), d_in[6] = k (fixed 16)
    const int B = in_sizes[0] / DIM;   // 4096
    const int M = in_sizes[3] / DIM;   // 65536

    dim3 grid(B / ROWS), block(256);
    sim_topk_kernel<<<grid, block, 0, stream>>>(features, cls, bank, bcls,
                                                (float*)d_out, B, M);
}
